// ShiftReduceModel_89206470738181
// MI455X (gfx1250) — compile-verified, matrix operand order checked by reading, not run on hardware
//
#include <hip/hip_runtime.h>
#include <hip/hip_bf16.h>

// ---------------- constants ----------------
#define NEDU  1024
#define LTOK  20
#define W2V   300
#define POSD  30
#define CW    330     // conv input width
#define CWP   352     // padded to 11 * 32 (bf16 K-tiles of 32 never straddle a tap)
#define NF    100
#define HD    256
#define FEAT  930
#define FEATP 960     // padded to 30 * 32
#define NSTEP 2047    // 2*NEDU - 1
#define NKT_TOT 66    // 11 + 22 + 33 K-tiles for widths 1,2,3
#define NT_CONV 7     // 112 = 7*16 filter tiles (100 used)

typedef __attribute__((ext_vector_type(16))) __bf16 v16bf;
typedef __attribute__((ext_vector_type(8)))  __bf16 v8bf;
typedef __attribute__((ext_vector_type(8)))  float  v8f;

__device__ __forceinline__ unsigned short f2bf(float f) {
    union { float f; unsigned int u; } v; v.f = f;
    unsigned int u = v.u;
    unsigned int r = u + 0x7fffu + ((u >> 16) & 1u);   // RNE
    return (unsigned short)(r >> 16);
}
__device__ __forceinline__ __bf16 us2bf(unsigned short s) {
    union { unsigned short s; __bf16 b; } u; u.s = s; return u.b;
}
__device__ __forceinline__ float sigm(float x) { return 1.f / (1.f + __expf(-x)); }

// K index inside a 32-K tile for WMMA 16-bit A/B fragment element j (0..15)
// lane<16 holds K {0..7,16..23}; lane>=16 holds K {8..15,24..31}
__device__ __forceinline__ int frag_k(int j, int half) {
    return (j < 8) ? (half * 8 + j) : (16 + half * 8 + (j - 8));
}

// ---------------- weight packing: conv filters -> B-fragment layout ----------------
// Bpack[kt][nt][lane][j], kt 0..65 (w1:0-10, w2:11-32, w3:33-65)
__global__ void pack_conv_w(const float* __restrict__ Wu, const float* __restrict__ Wb,
                            const float* __restrict__ Wt, unsigned short* __restrict__ Bpack) {
    int e = blockIdx.x * blockDim.x + threadIdx.x;
    const int total = NKT_TOT * NT_CONV * 32 * 16;
    if (e >= total) return;
    int j    = e & 15;
    int lane = (e >> 4) & 31;
    int nt   = (e >> 9) % NT_CONV;
    int kt   = e / (16 * 32 * NT_CONV);
    int w, kt2;
    if (kt < 11)      { w = 0; kt2 = kt; }
    else if (kt < 33) { w = 1; kt2 = kt - 11; }
    else              { w = 2; kt2 = kt - 33; }
    int half = lane >> 4;
    int kl   = kt2 * 32 + frag_k(j, half);     // kappa within k*352
    int dt   = kl / CWP, c = kl % CWP;
    int f    = nt * 16 + (lane & 15);
    float val = 0.f;
    if (f < NF && c < CW) {
        const float* W = (w == 0) ? Wu : (w == 1) ? Wb : Wt;
        int kw = w + 1;
        val = W[(f * kw + dt) * CW + c];
    }
    Bpack[e] = f2bf(val);
}

// ---------------- weight packing: Wp (512 x 930) -> B-fragment layout ----------------
// WpPack[kt(30)][nt(32)][lane][j]
__global__ void pack_wp(const float* __restrict__ Wp, unsigned short* __restrict__ WpPack) {
    int e = blockIdx.x * blockDim.x + threadIdx.x;
    const int total = 30 * 32 * 32 * 16;
    if (e >= total) return;
    int j    = e & 15;
    int lane = (e >> 4) & 31;
    int nt   = (e >> 9) & 31;
    int kt   = e >> 14;
    int half = lane >> 4;
    int kk   = kt * 32 + frag_k(j, half);
    int outc = nt * 16 + (lane & 15);
    float val = (kk < FEAT) ? Wp[(size_t)outc * FEAT + kk] : 0.f;
    WpPack[e] = f2bf(val);
}

// ---------------- per-EDU encoder: embed gather + WMMA conv + relu/max + feats ----------------
__global__ __launch_bounds__(224) void encode_conv(
    const int* __restrict__ words, const int* __restrict__ poses,
    const float* __restrict__ wemb, const float* __restrict__ pemb,
    const unsigned short* __restrict__ Bpack,
    const float* __restrict__ bu, const float* __restrict__ bb, const float* __restrict__ bt,
    float* __restrict__ feats)
{
    __shared__ __align__(16) unsigned short sX[34 * CWP];   // rows 0,1 and 22..33 stay zero
    const int edu = blockIdx.x;
    const int tid = threadIdx.x;

    for (int i = tid; i < 34 * CWP; i += 224) sX[i] = 0;
    __syncthreads();

    const int* w = words + edu * LTOK;
    const int* p = poses + edu * LTOK;
    for (int i = tid; i < LTOK * CW; i += 224) {
        int t = i / CW, c = i % CW;
        float v = (c < W2V) ? wemb[(size_t)w[t] * W2V + c]
                            : pemb[p[t] * POSD + (c - W2V)];
        sX[(t + 2) * CWP + c] = f2bf(v);
    }
    float* fr = feats + (size_t)edu * FEATP;
    for (int i = tid; i < 630; i += 224) {
        float v;
        if (i < 300)      v = wemb[(size_t)w[0] * W2V + i];
        else if (i < 600) v = wemb[(size_t)w[LTOK - 1] * W2V + (i - 300)];
        else              v = pemb[p[0] * POSD + (i - 600)];
        fr[i] = v;
    }
    for (int i = FEAT + tid; i < FEATP; i += 224) fr[i] = 0.f;
    __syncthreads();

    const int lane = tid & 31;
    const int nt   = tid >> 5;            // wave id = filter N-tile (0..6)
    const int half = lane >> 4;
    const int fcol = nt * 16 + (lane & 15);

    for (int w3 = 0; w3 < 3; ++w3) {      // width = w3+1, pad = w3
        float lmax = -1e30f;
        const int outlen = 20 + w3;
        const int ktbase = (w3 == 0) ? 0 : (w3 == 1) ? 11 : 33;
        const int nkt = 11 * (w3 + 1);
        for (int mt = 0; mt < 2; ++mt) {
            v8f acc = {};
            for (int kt2 = 0; kt2 < nkt; ++kt2) {
                int dt  = kt2 / 11;
                int cb  = (kt2 % 11) * 32;
                int row = mt * 16 + (lane & 15) + dt + 2 - w3;   // A row (time), in [0,33]
                const v8bf* a0 = (const v8bf*)(sX + row * CWP + cb + half * 8);
                const v8bf* a1 = (const v8bf*)(sX + row * CWP + cb + 16 + half * 8);
                v8bf alo = *a0, ahi = *a1;
                v16bf a = __builtin_shufflevector(alo, ahi,
                          0,1,2,3,4,5,6,7,8,9,10,11,12,13,14,15);
                const v16bf* bf = (const v16bf*)(Bpack +
                          (((size_t)(ktbase + kt2) * NT_CONV + nt) * 32 + lane) * 16);
                acc = __builtin_amdgcn_wmma_f32_16x16x32_bf16(
                          false, a, false, *bf, (short)0, acc, false, false);
            }
#pragma unroll
            for (int vg = 0; vg < 8; ++vg) {          // D: lane half h holds M = vg + 8h
                int o = mt * 16 + vg + half * 8;
                if (o < outlen) lmax = fmaxf(lmax, acc[vg]);
            }
        }
        lmax = fmaxf(lmax, __shfl_xor(lmax, 16));      // combine the two M-halves
        if (fcol < NF && half == 0) {
            const float* bias = (w3 == 0) ? bu : (w3 == 1) ? bb : bt;
            fr[630 + w3 * NF + fcol] = fmaxf(lmax + bias[fcol], 0.f);   // relu(max+b)
        }
    }
}

// ---------------- projection: feats[1024,960] x WpPack -> buf_h/buf_c ----------------
__global__ __launch_bounds__(256) void project_wmma(
    const float* __restrict__ feats, const unsigned short* __restrict__ WpPack,
    const float* __restrict__ bpv, float* __restrict__ buf_h, float* __restrict__ buf_c)
{
    const int mt   = blockIdx.x;          // 0..63 (M tiles of 16 EDUs)
    const int tid  = threadIdx.x;
    const int lane = tid & 31, wave = tid >> 5, half = lane >> 4;
    const int arow = mt * 16 + (lane & 15);
    const float* fr = feats + (size_t)arow * FEATP;

    for (int nn = 0; nn < 4; ++nn) {
        const int nt = wave + nn * 8;     // 0..31 (N tiles of 16 outputs)
        v8f acc = {};
        for (int kt = 0; kt < 30; ++kt) {
            int kb = kt * 32 + half * 8;
            const float4* q0 = (const float4*)(fr + kb);
            const float4* q1 = (const float4*)(fr + kb + 16);
            float4 x0 = q0[0], x1 = q0[1], x2 = q1[0], x3 = q1[1];
            float tmp[16] = { x0.x,x0.y,x0.z,x0.w, x1.x,x1.y,x1.z,x1.w,
                              x2.x,x2.y,x2.z,x2.w, x3.x,x3.y,x3.z,x3.w };
            v16bf a;
#pragma unroll
            for (int i = 0; i < 16; ++i) a[i] = us2bf(f2bf(tmp[i]));
            const v16bf* bf = (const v16bf*)(WpPack +
                              (((size_t)kt * 32 + nt) * 32 + lane) * 16);
            acc = __builtin_amdgcn_wmma_f32_16x16x32_bf16(
                      false, a, false, *bf, (short)0, acc, false, false);
        }
#pragma unroll
        for (int vg = 0; vg < 8; ++vg) {
            int row = mt * 16 + vg + half * 8;
            int col = nt * 16 + (lane & 15);
            float v = acc[vg] + bpv[col];
            if (col < HD) buf_h[(size_t)row * HD + col]        = v;
            else          buf_c[(size_t)row * HD + (col - HD)] = v;
        }
    }
}

// ---------------- serial shift-reduce scan (single persistent workgroup) ----------------
__device__ __forceinline__ void track_step(
    int tid, int sptr, int bptr,
    const float* __restrict__ Wih, const float* __restrict__ Whh,
    const float* __restrict__ bih, const float* __restrict__ bhh,
    const float* __restrict__ stack_h, const float* __restrict__ buf_h,
    float* s_xv, float* s_g, float* s_th, float* s_tc)
{
    if (tid < 3 * HD) {
        float v;
        if (tid < HD)          v = stack_h[(size_t)(sptr - 1) * HD + tid];
        else if (tid < 2 * HD) v = stack_h[(size_t)(sptr - 2) * HD + (tid - HD)];
        else                   v = buf_h[(size_t)bptr * HD + (tid - 2 * HD)];
        s_xv[tid] = v;
    }
    __syncthreads();
    {   // g[tid] for tid in [0,1024): Wih row dot xv + Whh row dot th + biases
        float acc = bih[tid] + bhh[tid];
        const float4* wr = (const float4*)(Wih + (size_t)tid * (3 * HD));
#pragma unroll 4
        for (int i = 0; i < (3 * HD) / 4; ++i) {
            float4 wv = wr[i]; int b = 4 * i;
            acc += wv.x * s_xv[b] + wv.y * s_xv[b + 1] + wv.z * s_xv[b + 2] + wv.w * s_xv[b + 3];
        }
        const float4* hr = (const float4*)(Whh + (size_t)tid * HD);
#pragma unroll 4
        for (int i = 0; i < HD / 4; ++i) {
            float4 wv = hr[i]; int b = 4 * i;
            acc += wv.x * s_th[b] + wv.y * s_th[b + 1] + wv.z * s_th[b + 2] + wv.w * s_th[b + 3];
        }
        s_g[tid] = acc;
    }
    __syncthreads();
    if (tid < HD) {   // torch LSTMCell gate order i,f,g,o
        float ig = s_g[tid], fg = s_g[HD + tid], gg = s_g[2 * HD + tid], og = s_g[3 * HD + tid];
        float c2 = sigm(fg) * s_tc[tid] + sigm(ig) * tanhf(gg);
        s_tc[tid] = c2;
        s_th[tid] = sigm(og) * tanhf(c2);
    }
    __syncthreads();
}

__global__ __launch_bounds__(1024) void scan_kernel(
    const int* __restrict__ transes,
    const float* __restrict__ Wih, const float* __restrict__ Whh,
    const float* __restrict__ bih, const float* __restrict__ bhh,
    const float* __restrict__ Wc,  const float* __restrict__ bc,
    const float* __restrict__ Ws,  const float* __restrict__ bs,
    float* __restrict__ buf_h, float* __restrict__ buf_c,
    float* __restrict__ stack_h, float* __restrict__ stack_c,
    float* __restrict__ out)
{
    __shared__ float s_th[HD], s_tc[HD], s_xv[3 * HD], s_g[5 * HD];
    __shared__ float s_c1[HD], s_c2[HD], s_log[2];
    const int tid = threadIdx.x;

    for (int i = tid; i < (NEDU + 2) * HD; i += 1024) { stack_h[i] = 0.f; stack_c[i] = 0.f; }
    for (int i = tid; i < 2 * HD; i += 1024) { buf_h[NEDU * HD + i] = 0.f; buf_c[NEDU * HD + i] = 0.f; }
    if (tid < HD) { s_th[tid] = 0.f; s_tc[tid] = 0.f; }
    __syncthreads();

    int sptr = 2, bptr = 0;
    track_step(tid, sptr, bptr, Wih, Whh, bih, bhh, stack_h, buf_h, s_xv, s_g, s_th, s_tc);

    float loss = 0.f;
    for (int step = 0; step < NSTEP; ++step) {
        const int trans = transes[step];          // uniform across the block
        if (tid < 2) {
            float acc = bs[tid];
            const float* wr = Ws + tid * HD;
            for (int i = 0; i < HD; ++i) acc += wr[i] * s_th[i];
            s_log[tid] = acc;
        }
        __syncthreads();
        if (tid == 0) {
            float l0 = s_log[0], l1 = s_log[1];
            float m = fmaxf(l0, l1);
            float lse = m + __logf(__expf(l0 - m) + __expf(l1 - m));
            loss += lse - (trans == 0 ? l0 : l1);
        }
        if (trans == 0) {                          // SHIFT
            if (tid < HD) {
                stack_h[(size_t)sptr * HD + tid] = buf_h[(size_t)bptr * HD + tid];
                stack_c[(size_t)sptr * HD + tid] = buf_c[(size_t)bptr * HD + tid];
            }
            sptr += 1; bptr += 1;
        } else {                                   // REDUCE (tree-LSTM)
            if (tid < 3 * HD) {
                float v;
                if (tid < HD)          v = stack_h[(size_t)(sptr - 1) * HD + tid];
                else if (tid < 2 * HD) v = stack_h[(size_t)(sptr - 2) * HD + (tid - HD)];
                else                   v = s_th[tid - 2 * HD];
                s_xv[tid] = v;
            }
            if (tid < 2 * HD) {
                if (tid < HD) s_c1[tid]      = stack_c[(size_t)(sptr - 1) * HD + tid];
                else          s_c2[tid - HD] = stack_c[(size_t)(sptr - 2) * HD + (tid - HD)];
            }
            __syncthreads();
            for (int j = tid; j < 5 * HD; j += 1024) {
                float acc = bc[j];
                const float4* wr = (const float4*)(Wc + (size_t)j * (3 * HD));
#pragma unroll 4
                for (int i = 0; i < (3 * HD) / 4; ++i) {
                    float4 wv = wr[i]; int b = 4 * i;
                    acc += wv.x * s_xv[b] + wv.y * s_xv[b + 1] + wv.z * s_xv[b + 2] + wv.w * s_xv[b + 3];
                }
                s_g[j] = acc;
            }
            __syncthreads();
            if (tid < HD) {   // gate order a,i,f1,f2,o
                float a  = s_g[tid],          ig = s_g[HD + tid];
                float f1 = s_g[2 * HD + tid], f2 = s_g[3 * HD + tid], og = s_g[4 * HD + tid];
                float rc = tanhf(a) * sigm(ig) + sigm(f1) * s_c1[tid] + sigm(f2) * s_c2[tid];
                float rh = sigm(og) * tanhf(rc);
                stack_h[(size_t)(sptr - 2) * HD + tid] = rh;
                stack_c[(size_t)(sptr - 2) * HD + tid] = rc;
            }
            sptr -= 1;
        }
        __syncthreads();
        track_step(tid, sptr, bptr, Wih, Whh, bih, bhh, stack_h, buf_h, s_xv, s_g, s_th, s_tc);
    }
    if (tid == 0) out[0] = loss / (float)NSTEP;
}

// ---------------- host entry ----------------
extern "C" void kernel_launch(void* const* d_in, const int* in_sizes, int n_in,
                              void* d_out, int out_size, void* d_ws, size_t ws_size,
                              hipStream_t stream) {
    const int*   edu_words = (const int*)d_in[0];
    const int*   edu_poses = (const int*)d_in[1];
    const int*   transes   = (const int*)d_in[2];
    const float* word_emb  = (const float*)d_in[3];
    const float* pos_emb   = (const float*)d_in[4];
    const float* Wu = (const float*)d_in[5];  const float* bu = (const float*)d_in[6];
    const float* Wb = (const float*)d_in[7];  const float* bb = (const float*)d_in[8];
    const float* Wt = (const float*)d_in[9];  const float* bt = (const float*)d_in[10];
    const float* Wp = (const float*)d_in[11]; const float* bp = (const float*)d_in[12];
    const float* Wih = (const float*)d_in[13]; const float* Whh = (const float*)d_in[14];
    const float* bih = (const float*)d_in[15]; const float* bhh = (const float*)d_in[16];
    const float* Wc  = (const float*)d_in[17]; const float* bc  = (const float*)d_in[18];
    const float* Ws  = (const float*)d_in[19]; const float* bs  = (const float*)d_in[20];

    char* ws = (char*)d_ws;
    size_t off = 0;
    auto alloc = [&](size_t bytes) -> char* {
        off = (off + 255) & ~(size_t)255;
        char* r = ws + off;
        off += bytes;
        return r;
    };
    unsigned short* Bpack  = (unsigned short*)alloc((size_t)NKT_TOT * NT_CONV * 32 * 16 * 2);
    unsigned short* WpPack = (unsigned short*)alloc((size_t)30 * 32 * 32 * 16 * 2);
    float* feats   = (float*)alloc((size_t)NEDU * FEATP * 4);
    float* buf_h   = (float*)alloc((size_t)(NEDU + 2) * HD * 4);
    float* buf_c   = (float*)alloc((size_t)(NEDU + 2) * HD * 4);
    float* stack_h = (float*)alloc((size_t)(NEDU + 2) * HD * 4);
    float* stack_c = (float*)alloc((size_t)(NEDU + 2) * HD * 4);

    {
        int tot = NKT_TOT * NT_CONV * 32 * 16;
        pack_conv_w<<<(tot + 255) / 256, 256, 0, stream>>>(Wu, Wb, Wt, Bpack);
    }
    {
        int tot = 30 * 32 * 32 * 16;
        pack_wp<<<(tot + 255) / 256, 256, 0, stream>>>(Wp, WpPack);
    }
    encode_conv<<<NEDU, 224, 0, stream>>>(edu_words, edu_poses, word_emb, pos_emb,
                                          Bpack, bu, bb, bt, feats);
    project_wmma<<<64, 256, 0, stream>>>(feats, WpPack, bp, buf_h, buf_c);
    scan_kernel<<<1, 1024, 0, stream>>>(transes, Wih, Whh, bih, bhh, Wc, bc, Ws, bs,
                                        buf_h, buf_c, stack_h, stack_c, (float*)d_out);
}
